// NTC_317827580318
// MI455X (gfx1250) — matrix operand order repaired and checked by reading, not compiled, not run on hardware
//
#include <hip/hip_runtime.h>
#include <hip/hip_bf16.h>

typedef __attribute__((ext_vector_type(16))) __bf16          v16bf;
typedef __attribute__((ext_vector_type(16))) unsigned short  v16us;
typedef __attribute__((ext_vector_type(8)))  float           v8f;

// float -> bf16 bits via native conversion (RTNE); compiler may use
// v_cvt_pk_bf16_f32-class hardware conversion on gfx1250.
__device__ __forceinline__ unsigned short f2bf(float x) {
    return __builtin_bit_cast(unsigned short, (__bf16)x);
}

// Map a logical im2col element (row m in 0..63, col kk in 0..255) to its
// slot in the pre-swizzled WMMA A-fragment buffer.
// A-matrix 16x32 bf16 layout (ISA 7.12.2): lanes 0-15 hold row M=mr with
// K in {0..7, 16..23}; lanes 16-31 hold the same rows with K in {8..15, 24..31}.
// Fragment buffer: [mtile(4)][ktile(8)][lane(32)][elem(16)] ushorts, so each
// lane's 16 elements are one contiguous 32B chunk (ds_load_b128 x2).
__device__ __forceinline__ int frag_index(int m, int kk) {
    int mt = m >> 4, mr = m & 15;
    int kt = kk >> 5, ko = kk & 31;
    int hi = (ko >> 3) & 1;                       // lane-half select
    int ei = (ko & 7) | (((ko >> 4) & 1) << 3);   // element within lane
    return (((((mt << 3) | kt) << 5) | (mr | (hi << 4))) << 4) | ei;
}

__launch_bounds__(256, 1)
__global__ void ntc_fused_kernel(const int* __restrict__ i_in,
                                 const int* __restrict__ j_in,
                                 const int* __restrict__ k_in,
                                 const float* __restrict__ user_emb,
                                 const float* __restrict__ item_emb,
                                 const float* __restrict__ time_emb,
                                 const float* __restrict__ w1g,
                                 const float* __restrict__ b1g,
                                 const float* __restrict__ w2g,
                                 const float* __restrict__ b2g,
                                 const float* __restrict__ w3g,
                                 const float* __restrict__ b3g,
                                 const float* __restrict__ fcwg,
                                 const float* __restrict__ fcbg,
                                 float* __restrict__ out) {
    // A-fragment buffers in pre-swizzled WMMA layout (64 rows x 256 cols each)
    __shared__ __align__(32) unsigned short A2[64 * 256];   // conv2 im2col (one slab)
    __shared__ __align__(32) unsigned short A3[64 * 256];   // conv3 im2col (full)
    __shared__ float xi_s[32], xj_s[32], xk_s[32];
    __shared__ float w1_s[256], b1_s[32];
    __shared__ float red[8];

    const int t    = threadIdx.x;
    const int lane = t & 31;
    const int wv   = t >> 5;          // wave 0..7
    const int hi16 = (lane >> 4) & 1; // lane half
    const int s    = blockIdx.x;      // sample

    // ---- stage embeddings + conv1 weights ----
    {
        const int iu = i_in[s], ij = j_in[s], ik = k_in[s];
        if (t < 32) {
            xi_s[t] = user_emb[iu * 32 + t];
            xj_s[t] = item_emb[ij * 32 + t];
            xk_s[t] = time_emb[ik * 32 + t];
            b1_s[t] = b1g[t];
        }
        w1_s[t] = w1g[t];  // 256 floats, 256 threads
    }

    // ---- per-wave GEMM tile assignment (same for conv2 and conv3) ----
    const int mt = wv >> 1;                       // M-tile 0..3
    const int cn = ((wv & 1) << 4) | (lane & 15); // output channel 0..31

    // ---- preload B fragments for conv2 (w2 reorg: B[kk][n] = w2[n*256+kk]) ----
    // B-matrix 32x16 bf16 layout: col N=lane&15; lane<16 -> K in {0..7,16..23}; lane>=16 -> K in {8..15,24..31}.
    v16us breg2[8];
#pragma unroll
    for (int kt = 0; kt < 8; ++kt) {
        const float* wp = &w2g[cn * 256 + kt * 32 + hi16 * 8];
#pragma unroll
        for (int ei = 0; ei < 16; ++ei) breg2[kt][ei] = f2bf(wp[ei + ((ei >> 3) << 3)]);
    }
    const float bias2 = b2g[cn];

    __syncthreads();

    // ---- hoisted conv1 partials: depend only on this thread's fixed (j,k) ----
    // s0a[c] = sum_{b,e} w1[c,0,0,b,e] * xj[2j+b]*xk[2k+e]   (i-tap a'=0)
    // s1a[c] = sum_{b,e} w1[c,0,1,b,e] * xj[2j+b]*xk[2k+e]   (i-tap a'=1)
    float s0a[32], s1a[32];
    {
        const int j = t >> 4, k = t & 15;
        const float p00 = xj_s[2 * j]     * xk_s[2 * k];
        const float p01 = xj_s[2 * j]     * xk_s[2 * k + 1];
        const float p10 = xj_s[2 * j + 1] * xk_s[2 * k];
        const float p11 = xj_s[2 * j + 1] * xk_s[2 * k + 1];
#pragma unroll
        for (int c = 0; c < 32; ++c) {
            const float* w = &w1_s[c * 8];
            s0a[c] = w[0] * p00 + w[1] * p01 + w[2] * p10 + w[3] * p11;
            s1a[c] = w[4] * p00 + w[5] * p01 + w[6] * p10 + w[7] * p11;
        }
    }

    // ---- slab loop over conv2 output dim di (conv1 fused in) ----
    for (int di = 0; di < 8; ++di) {
        // conv1: fill A2 with y1 planes i = 2*di, 2*di+1 (non-overlapping patches
        // => direct scatter into im2col fragment slots)
        {
            const int j = t >> 4, k = t & 15;
            const float xi0 = xi_s[4 * di], xi1 = xi_s[4 * di + 1];
            const float xi2 = xi_s[4 * di + 2], xi3 = xi_s[4 * di + 3];
            const int m   = ((j >> 1) << 3) | (k >> 1);
            const int kbe = (((j & 1) << 1) | (k & 1));
#pragma unroll
            for (int c = 0; c < 32; ++c) {
                float v0 = b1_s[c] + xi0 * s0a[c] + xi1 * s1a[c];  // a = 0
                float v1 = b1_s[c] + xi2 * s0a[c] + xi3 * s1a[c];  // a = 1
                v0 = fmaxf(v0, 0.f);
                v1 = fmaxf(v1, 0.f);
                A2[frag_index(m, c * 8 + kbe)]     = f2bf(v0);
                A2[frag_index(m, c * 8 + 4 + kbe)] = f2bf(v1);
            }
        }
        __syncthreads();

        // conv2 GEMM: 16x16 tile per wave, K = 256 in 8 WMMA steps
        v8f acc = {0.f, 0.f, 0.f, 0.f, 0.f, 0.f, 0.f, 0.f};
#pragma unroll
        for (int kt = 0; kt < 8; ++kt) {
            const v16us au = *(const v16us*)&A2[(((((mt << 3) | kt) << 5) | lane) << 4)];
            acc = __builtin_amdgcn_wmma_f32_16x16x32_bf16(
                false, __builtin_bit_cast(v16bf, au),
                false, __builtin_bit_cast(v16bf, breg2[kt]),
                (short)0, acc, false, false);
        }

        // epilogue: bias + ReLU, scatter into conv3 im2col fragment slots.
        // C/D layout: VGPR r -> M = r + 8*(lane>=16), N = lane&15 within tile.
#pragma unroll
        for (int r = 0; r < 8; ++r) {
            const int m  = (mt << 4) + r + (hi16 << 3);  // position in slab (dj*8+dk)
            const int dj = m >> 3, dk = m & 7;
            const float val = fmaxf(acc[r] + bias2, 0.f);
            const int m3  = ((di >> 1) << 4) | ((dj >> 1) << 2) | (dk >> 1);
            const int kk3 = cn * 8 + ((di & 1) << 2) + ((dj & 1) << 1) + (dk & 1);
            A3[frag_index(m3, kk3)] = f2bf(val);
        }
        __syncthreads();
    }

    // ---- preload B fragments for conv3 ----
    v16us breg3[8];
#pragma unroll
    for (int kt = 0; kt < 8; ++kt) {
        const float* wp = &w3g[cn * 256 + kt * 32 + hi16 * 8];
#pragma unroll
        for (int ei = 0; ei < 16; ++ei) breg3[kt][ei] = f2bf(wp[ei + ((ei >> 3) << 3)]);
    }
    const float bias3 = b3g[cn];

    // ---- conv3 GEMM: M=64, K=256, N=32 ----
    v8f acc3 = {0.f, 0.f, 0.f, 0.f, 0.f, 0.f, 0.f, 0.f};
#pragma unroll
    for (int kt = 0; kt < 8; ++kt) {
        const v16us au = *(const v16us*)&A3[(((((mt << 3) | kt) << 5) | lane) << 4)];
        acc3 = __builtin_amdgcn_wmma_f32_16x16x32_bf16(
            false, __builtin_bit_cast(v16bf, au),
            false, __builtin_bit_cast(v16bf, breg3[kt]),
            (short)0, acc3, false, false);
    }

    // ---- FC partial: flatten index f = c*64 + m3; fc_w reads are contiguous in r ----
    float partial = 0.f;
    {
        const int mbase = (mt << 4) + (hi16 << 3);
        const float* fp = &fcwg[cn * 64 + mbase];
#pragma unroll
        for (int r = 0; r < 8; ++r) {
            const float val = fmaxf(acc3[r] + bias3, 0.f);
            partial += val * fp[r];
        }
    }

    // ---- reduce 256 threads -> scalar, sigmoid, store ----
#pragma unroll
    for (int off = 16; off > 0; off >>= 1)
        partial += __shfl_down(partial, off, 32);
    if (lane == 0) red[wv] = partial;
    __syncthreads();
    if (t == 0) {
        float acc_s = fcbg[0];
#pragma unroll
        for (int w = 0; w < 8; ++w) acc_s += red[w];
        out[s] = 1.f / (1.f + expf(-acc_s));
    }
}

extern "C" void kernel_launch(void* const* d_in, const int* in_sizes, int n_in,
                              void* d_out, int out_size, void* d_ws, size_t ws_size,
                              hipStream_t stream) {
    const int*   i_in     = (const int*)d_in[0];
    const int*   j_in     = (const int*)d_in[1];
    const int*   k_in     = (const int*)d_in[2];
    const float* user_emb = (const float*)d_in[3];
    const float* item_emb = (const float*)d_in[4];
    const float* time_emb = (const float*)d_in[5];
    const float* w1       = (const float*)d_in[6];
    const float* b1       = (const float*)d_in[7];
    const float* w2       = (const float*)d_in[8];
    const float* b2       = (const float*)d_in[9];
    const float* w3       = (const float*)d_in[10];
    const float* b3       = (const float*)d_in[11];
    const float* fc_w     = (const float*)d_in[12];
    const float* fc_b     = (const float*)d_in[13];
    float* out = (float*)d_out;

    const int B = in_sizes[0];  // 1024 samples, one workgroup each
    ntc_fused_kernel<<<B, 256, 0, stream>>>(i_in, j_in, k_in,
                                            user_emb, item_emb, time_emb,
                                            w1, b1, w2, b2, w3, b3,
                                            fc_w, fc_b, out);
}
